// MultiHeadAttention_35905926595104
// MI455X (gfx1250) — matrix operand order repaired and checked by reading, not compiled, not run on hardware
//
#include <hip/hip_runtime.h>
#include <math.h>

// ---------------------------------------------------------------------------
// Types for CDNA5 WMMA (wave32, 16x16x32 bf16 -> f32)
// ---------------------------------------------------------------------------
typedef __attribute__((ext_vector_type(16))) __bf16 v16bf;
typedef __attribute__((ext_vector_type(8)))  __bf16 v8bf;
typedef __attribute__((ext_vector_type(4)))  __bf16 v4bf;
typedef __attribute__((ext_vector_type(8)))  float  v8f;

#define NB_  2
#define NH_  12
#define HD_  64
#define EMB_ 768
#define QL_  1024
#define KL_  2048
#define SCALE_  0.125f       /* 1/sqrt(64) */
#define NEG_BIG (-1e9f)
#define NEG_INF (-3.0e38f)

static __device__ __forceinline__ v8f vzero8() {
  v8f z;
#pragma unroll
  for (int i = 0; i < 8; ++i) z[i] = 0.f;
  return z;
}

static __device__ __forceinline__ v16bf combine16(v8bf lo, v8bf hi) {
  v16bf r;
#pragma unroll
  for (int i = 0; i < 8; ++i) { r[i] = lo[i]; r[i + 8] = hi[i]; }
  return r;
}

// A-fragment (16x32 bf16). Lane L holds row m; half i maps to
// K = (L>>4)*16 + i  => 16 contiguous K values.
static __device__ __forceinline__ v16bf ld_afrag(const __bf16* rowp, int lane) {
  const int g = (lane >> 4) << 4;
  v8bf lo = *(const v8bf*)(rowp + g);
  v8bf hi = *(const v8bf*)(rowp + g + 8);
  return combine16(lo, hi);
}

// B-fragment (32x16 bf16). Lane L holds col n = L&15; 16 contiguous K values
// starting at (L>>4)*16. Caller passes the per-lane start pointer.
static __device__ __forceinline__ v16bf ld_bfrag(const __bf16* p) {
  v8bf lo = *(const v8bf*)p;
  v8bf hi = *(const v8bf*)(p + 8);
  return combine16(lo, hi);
}

static __device__ __forceinline__ v8f wmma_bf16f32(v16bf a, v16bf b, v8f c) {
  return __builtin_amdgcn_wmma_f32_16x16x32_bf16(false, a, false, b,
                                                 (short)0, c, false, false);
}

// ---------------------------------------------------------------------------
// Generic row-major GEMM: C[M x 768] = A[M x 768] * W[768 x 768] (+ bias)
// Block 128x128, 8 waves (4x2), wave tile 32x64, fp32->bf16 conversion in LDS.
// M must be a multiple of 128 (true for all call sites).
// OMODE: 0 = fp32 row-major, 1 = bf16 row-major,
//        2 = bf16 transposed "[b][col][k]" (k-major, for attention V operand;
//            rows are interpreted as b*KL + k)
// ---------------------------------------------------------------------------
template <int OMODE>
__global__ __launch_bounds__(256) void gemm_rowmajor(
    const float* __restrict__ A, const float* __restrict__ W,
    const float* __restrict__ bias, float* __restrict__ Cf,
    __bf16* __restrict__ Cb) {
  __shared__ __attribute__((aligned(16))) __bf16 As[128 * 40];
  __shared__ __attribute__((aligned(16))) __bf16 Bs[128 * 40];  // [n][k]

  const int tid = threadIdx.x;
  const int lane = tid & 31;
  const int w = tid >> 5;
  const int wm = w >> 1, wn = w & 1;
  const int rowBase = blockIdx.x * 128;
  const int colBase = blockIdx.y * 128;

  v8f acc[2][4];
#pragma unroll
  for (int i = 0; i < 2; ++i)
#pragma unroll
    for (int j = 0; j < 4; ++j) acc[i][j] = vzero8();

  for (int k0 = 0; k0 < EMB_; k0 += 32) {
    __syncthreads();
#pragma unroll
    for (int i = 0; i < 4; ++i) {           // A tile: 128 x 32 fp32 -> bf16
      int cid = i * 256 + tid;
      int r = cid >> 3, c4 = (cid & 7) << 2;
      float4 v = *(const float4*)(A + (size_t)(rowBase + r) * EMB_ + k0 + c4);
      v4bf bv;
      bv[0] = (__bf16)v.x; bv[1] = (__bf16)v.y;
      bv[2] = (__bf16)v.z; bv[3] = (__bf16)v.w;
      *(v4bf*)&As[r * 40 + c4] = bv;
    }
#pragma unroll
    for (int i = 0; i < 4; ++i) {           // W tile: 32 x 128, transposed
      int cid = i * 256 + tid;
      int kr = cid >> 5, n4 = (cid & 31) << 2;
      float4 v = *(const float4*)(W + (size_t)(k0 + kr) * EMB_ + colBase + n4);
      Bs[(n4 + 0) * 40 + kr] = (__bf16)v.x;
      Bs[(n4 + 1) * 40 + kr] = (__bf16)v.y;
      Bs[(n4 + 2) * 40 + kr] = (__bf16)v.z;
      Bs[(n4 + 3) * 40 + kr] = (__bf16)v.w;
    }
    if (k0 + 32 < EMB_) {                   // hint next tiles into cache
      __builtin_prefetch(A + (size_t)(rowBase + (tid >> 1)) * EMB_ + k0 + 32, 0, 1);
      __builtin_prefetch(W + (size_t)(k0 + 32 + (tid & 31)) * EMB_ + colBase, 0, 1);
    }
    __syncthreads();

    v16bf afr[2], bfr[4];
#pragma unroll
    for (int fm = 0; fm < 2; ++fm)
      afr[fm] = ld_afrag(&As[(wm * 32 + fm * 16 + (lane & 15)) * 40], lane);
#pragma unroll
    for (int fn = 0; fn < 4; ++fn)
      bfr[fn] = ld_bfrag(&Bs[(wn * 64 + fn * 16 + (lane & 15)) * 40 +
                             ((lane >> 4) << 4)]);
#pragma unroll
    for (int fm = 0; fm < 2; ++fm)
#pragma unroll
      for (int fn = 0; fn < 4; ++fn)
        acc[fm][fn] = wmma_bf16f32(afr[fm], bfr[fn], acc[fm][fn]);
  }

#pragma unroll
  for (int fm = 0; fm < 2; ++fm)
#pragma unroll
    for (int fn = 0; fn < 4; ++fn) {
      int col = colBase + wn * 64 + fn * 16 + (lane & 15);
      float bv = bias ? bias[col] : 0.f;
#pragma unroll
      for (int r = 0; r < 8; ++r) {
        int row = rowBase + wm * 32 + fm * 16 + r + ((lane >> 4) << 3);
        float val = acc[fm][fn][r] + bv;
        if (OMODE == 0) {
          Cf[(size_t)row * EMB_ + col] = val;
        } else if (OMODE == 1) {
          Cb[(size_t)row * EMB_ + col] = (__bf16)val;
        } else {
          int bi = row >> 11;           // row / KL
          int kk = row & (KL_ - 1);     // row % KL
          Cb[((size_t)bi * EMB_ + col) * KL_ + kk] = (__bf16)val;
        }
      }
    }
}

// ---------------------------------------------------------------------------
// Attention pass 1: streaming softmax stats (row max + sum-exp).
// Block = (b, h, 32 q-rows). S tile = 32 x 128 computed with WMMA
// (AC + rel-shifted BD), masked+scaled exactly like the reference.
// ---------------------------------------------------------------------------
__global__ __launch_bounds__(256) void attn_stats(
    const float* __restrict__ qf, const __bf16* __restrict__ kbf,
    const __bf16* __restrict__ rbf, const float* __restrict__ rwb,
    const float* __restrict__ rrb, const unsigned char* __restrict__ amask,
    float* __restrict__ Mws, float* __restrict__ Lws) {
  const int qt = blockIdx.x, h = blockIdx.y, b = blockIdx.z;
  const int qbase = qt * 32;
  // Faithful _rel_shift head remap (flat reinterpretation over (b, h+1)).
  const int mm = b * NH_ + h + NB_;
  const int bb = mm / (NH_ + 1);
  const int hh = mm % (NH_ + 1);
  const bool zeroBD = (hh == 0);
  const int hs = zeroBD ? 0 : hh - 1;

  __shared__ __attribute__((aligned(16))) __bf16 qwL[32 * 72];
  __shared__ __attribute__((aligned(16))) __bf16 qrL[32 * 72];
  __shared__ float SL[32 * 132];
  __shared__ float red[256];
  __shared__ float tmx[32];
  __shared__ float rowm[32], rowl[32];

  const int tid = threadIdx.x, lane = tid & 31, w = tid >> 5;
  const int wr = w >> 2, wc = w & 3;

#pragma unroll
  for (int i = 0; i < 8; ++i) {            // stage q + biases as bf16
    int cid = i * 256 + tid;
    int r = cid >> 6, d = cid & 63;
    float qv = qf[((size_t)b * QL_ + qbase + r) * EMB_ + h * HD_ + d] +
               rwb[h * HD_ + d];
    qwL[r * 72 + d] = (__bf16)qv;
    if (!zeroBD) {
      float q2 = qf[((size_t)bb * QL_ + qbase + r) * EMB_ + hs * HD_ + d] +
                 rrb[hs * HD_ + d];
      qrL[r * 72 + d] = (__bf16)q2;
    }
  }
  if (tid < 32) { rowm[tid] = NEG_INF; rowl[tid] = 0.f; }
  __syncthreads();

  const __bf16* qwrow = &qwL[(wr * 16 + (lane & 15)) * 72];
  const __bf16* qrrow = &qrL[(wr * 16 + (lane & 15)) * 72];
  const int g16 = (lane >> 4) << 4;
  const int rrow = tid >> 3, chunk = tid & 7;

  for (int k0 = 0; k0 < KL_; k0 += 128) {
    v8f acc[2];
#pragma unroll
    for (int fn = 0; fn < 2; ++fn) {
      const int kpos = k0 + wc * 32 + fn * 16 + (lane & 15);
      v8f a = vzero8();
      const __bf16* kp = kbf + ((size_t)b * KL_ + kpos) * EMB_ + h * HD_;
#pragma unroll
      for (int dblk = 0; dblk < 2; ++dblk)
        a = wmma_bf16f32(ld_afrag(qwrow + dblk * 32, lane),
                         ld_bfrag(kp + dblk * 32 + g16), a);
      if (!zeroBD) {
        const __bf16* rp = rbf + (size_t)kpos * EMB_ + hs * HD_;
#pragma unroll
        for (int dblk = 0; dblk < 2; ++dblk)
          a = wmma_bf16f32(ld_afrag(qrrow + dblk * 32, lane),
                           ld_bfrag(rp + dblk * 32 + g16), a);
      }
      acc[fn] = a;
    }
    __syncthreads();
#pragma unroll
    for (int fn = 0; fn < 2; ++fn) {
      int col = wc * 32 + fn * 16 + (lane & 15);
#pragma unroll
      for (int r = 0; r < 8; ++r)
        SL[(wr * 16 + r + ((lane >> 4) << 3)) * 132 + col] = acc[fn][r];
    }
    __syncthreads();

    // scale + mask, then running (m, l) update: 8 threads per row
    float* srow = &SL[rrow * 132 + chunk * 16];
    const unsigned char* mrow =
        amask + ((size_t)b * QL_ + qbase + rrow) * KL_ + k0 + chunk * 16;
    float lm = NEG_INF;
#pragma unroll
    for (int j = 0; j < 16; ++j) {
      float s = mrow[j] ? NEG_BIG : srow[j] * SCALE_;
      srow[j] = s;
      lm = fmaxf(lm, s);
    }
    red[tid] = lm;
    __syncthreads();
    if (chunk == 0) {
      float t = red[tid];
#pragma unroll
      for (int j = 1; j < 8; ++j) t = fmaxf(t, red[tid + j]);
      tmx[rrow] = fmaxf(t, rowm[rrow]);
    }
    __syncthreads();
    const float mn = tmx[rrow];
    float ls = 0.f;
#pragma unroll
    for (int j = 0; j < 16; ++j) ls += __expf(srow[j] - mn);
    red[tid] = ls;
    __syncthreads();
    if (chunk == 0) {
      float t = 0.f;
#pragma unroll
      for (int j = 0; j < 8; ++j) t += red[tid + j];
      rowl[rrow] = rowl[rrow] * __expf(rowm[rrow] - mn) + t;
      rowm[rrow] = mn;
    }
    __syncthreads();
  }
  if (tid < 32) {
    size_t idx = ((size_t)(b * NH_ + h)) * QL_ + qbase + tid;
    Mws[idx] = rowm[tid];
    Lws[idx] = rowl[tid];
  }
}

// ---------------------------------------------------------------------------
// Attention pass 2: recompute S tiles, write normalized prob to HBM,
// accumulate O = P @ V with WMMA. P goes through LDS for the C->A relayout;
// V is consumed directly from its k-major global layout (B-fragment order),
// so no per-tile LDS transpose is needed.
// ---------------------------------------------------------------------------
__global__ __launch_bounds__(256) void attn_pv(
    const float* __restrict__ qf, const __bf16* __restrict__ kbf,
    const __bf16* __restrict__ rbf, const __bf16* __restrict__ vtb,
    const float* __restrict__ rwb, const float* __restrict__ rrb,
    const unsigned char* __restrict__ amask, const float* __restrict__ Mws,
    const float* __restrict__ Lws, float* __restrict__ prob,
    float* __restrict__ vec) {
  const int qt = blockIdx.x, h = blockIdx.y, b = blockIdx.z;
  const int qbase = qt * 32;
  const int mm = b * NH_ + h + NB_;
  const int bb = mm / (NH_ + 1);
  const int hh = mm % (NH_ + 1);
  const bool zeroBD = (hh == 0);
  const int hs = zeroBD ? 0 : hh - 1;

  __shared__ __attribute__((aligned(16))) __bf16 qwL[32 * 72];
  __shared__ __attribute__((aligned(16))) __bf16 qrL[32 * 72];
  __shared__ float SL[32 * 132];
  __shared__ __attribute__((aligned(16))) __bf16 PL[32 * 136];
  __shared__ float rowm[32], rinv[32];

  const int tid = threadIdx.x, lane = tid & 31, w = tid >> 5;
  const int wr = w >> 2, wc = w & 3;    // S-phase roles
  const int fm2 = w >> 2, fn4 = w & 3;  // PV-phase roles

#pragma unroll
  for (int i = 0; i < 8; ++i) {
    int cid = i * 256 + tid;
    int r = cid >> 6, d = cid & 63;
    float qv = qf[((size_t)b * QL_ + qbase + r) * EMB_ + h * HD_ + d] +
               rwb[h * HD_ + d];
    qwL[r * 72 + d] = (__bf16)qv;
    if (!zeroBD) {
      float q2 = qf[((size_t)bb * QL_ + qbase + r) * EMB_ + hs * HD_ + d] +
                 rrb[hs * HD_ + d];
      qrL[r * 72 + d] = (__bf16)q2;
    }
  }
  if (tid < 32) {
    size_t idx = ((size_t)(b * NH_ + h)) * QL_ + qbase + tid;
    rowm[tid] = Mws[idx];
    float l = Lws[idx];
    rinv[tid] = (l != 0.f) ? 1.f / l : 0.f;
  }

  const __bf16* qwrow = &qwL[(wr * 16 + (lane & 15)) * 72];
  const __bf16* qrrow = &qrL[(wr * 16 + (lane & 15)) * 72];
  const int g16 = (lane >> 4) << 4;
  // Per-lane base into k-major V: row = b*768 + h*64 + (fn4*16 + n-lane)
  const __bf16* vrow =
      vtb + ((size_t)b * EMB_ + h * HD_ + fn4 * 16 + (lane & 15)) * KL_ + g16;
  v8f oacc = vzero8();

  for (int k0 = 0; k0 < KL_; k0 += 128) {
    v8f acc[2];
#pragma unroll
    for (int fn = 0; fn < 2; ++fn) {
      const int kpos = k0 + wc * 32 + fn * 16 + (lane & 15);
      v8f a = vzero8();
      const __bf16* kp = kbf + ((size_t)b * KL_ + kpos) * EMB_ + h * HD_;
#pragma unroll
      for (int dblk = 0; dblk < 2; ++dblk)
        a = wmma_bf16f32(ld_afrag(qwrow + dblk * 32, lane),
                         ld_bfrag(kp + dblk * 32 + g16), a);
      if (!zeroBD) {
        const __bf16* rp = rbf + (size_t)kpos * EMB_ + hs * HD_;
#pragma unroll
        for (int dblk = 0; dblk < 2; ++dblk)
          a = wmma_bf16f32(ld_afrag(qrrow + dblk * 32, lane),
                           ld_bfrag(rp + dblk * 32 + g16), a);
      }
      acc[fn] = a;
    }
    __syncthreads();  // prior-iteration LDS consumers done
#pragma unroll
    for (int fn = 0; fn < 2; ++fn) {
      int col = wc * 32 + fn * 16 + (lane & 15);
#pragma unroll
      for (int r = 0; r < 8; ++r)
        SL[(wr * 16 + r + ((lane >> 4) << 3)) * 132 + col] = acc[fn][r];
    }
    __syncthreads();

    // p = exp(masked_scaled_s - m) / l ; write prob (HBM) + PL (bf16)
#pragma unroll
    for (int i = 0; i < 16; ++i) {
      int e = i * 256 + tid;
      int row = e >> 7, col = e & 127;
      float s = SL[row * 132 + col];
      unsigned char mk =
          amask[((size_t)b * QL_ + qbase + row) * KL_ + k0 + col];
      s = mk ? NEG_BIG : s * SCALE_;
      float p = __expf(s - rowm[row]) * rinv[row];
      prob[(((size_t)(b * NH_ + h)) * QL_ + qbase + row) * KL_ + k0 + col] = p;
      PL[row * 136 + col] = (__bf16)p;
    }
    __syncthreads();

    // O += P(32x128) @ V(128x64): each wave one 16x16 tile, K=128 -> 4 WMMA.
    // V B-fragments load straight from global (k-major layout).
    const __bf16* prow = &PL[(fm2 * 16 + (lane & 15)) * 136];
#pragma unroll
    for (int kk = 0; kk < 4; ++kk)
      oacc = wmma_bf16f32(ld_afrag(prow + kk * 32, lane),
                          ld_bfrag(vrow + k0 + kk * 32), oacc);
  }

#pragma unroll
  for (int r = 0; r < 8; ++r) {
    int row = qbase + fm2 * 16 + r + ((lane >> 4) << 3);
    int col = fn4 * 16 + (lane & 15);
    vec[((size_t)b * QL_ + row) * EMB_ + h * HD_ + col] = oacc[r];
  }
}

// ---------------------------------------------------------------------------
// Launch: 4 projection GEMMs -> stats pass -> prob+PV pass -> output GEMM.
// Workspace (bytes):
//   qf   fp32  [2,1024,768]        @ 0         (6291456)
//   kbf  bf16  [2,2048,768]        @ 6291456   (6291456)
//   vtb  bf16  [2,768,2048] k-major@ 12582912  (6291456)
//   rbf  bf16  [2048,768]          @ 18874368  (3145728)
//   Mws  fp32  [2,12,1024]         @ 22020096  (98304)
//   Lws  fp32  [2,12,1024]         @ 22118400  (98304)
//   vec  fp32  [2,1024,768]        @ 22216704  (6291456)   total ~28.5 MB
// ---------------------------------------------------------------------------
extern "C" void kernel_launch(void* const* d_in, const int* in_sizes, int n_in,
                              void* d_out, int out_size, void* d_ws,
                              size_t ws_size, hipStream_t stream) {
  (void)in_sizes; (void)n_in; (void)out_size; (void)ws_size;
  const float* Q   = (const float*)d_in[0];
  const float* K   = (const float*)d_in[1];
  const float* V   = (const float*)d_in[2];
  const float* pos = (const float*)d_in[3];
  const float* rwb = (const float*)d_in[4];
  const float* rrb = (const float*)d_in[5];
  const unsigned char* amask = (const unsigned char*)d_in[6];
  const float* Wq = (const float*)d_in[8];
  const float* bq = (const float*)d_in[9];
  const float* Wk = (const float*)d_in[10];
  const float* bk = (const float*)d_in[11];
  const float* Wv = (const float*)d_in[12];
  const float* bv = (const float*)d_in[13];
  const float* Wr = (const float*)d_in[14];
  const float* Wo = (const float*)d_in[15];
  const float* bo = (const float*)d_in[16];

  char* ws = (char*)d_ws;
  float*  qf  = (float*)(ws);
  __bf16* kbf = (__bf16*)(ws + 6291456);
  __bf16* vtb = (__bf16*)(ws + 12582912);
  __bf16* rbf = (__bf16*)(ws + 18874368);
  float*  Mws = (float*)(ws + 22020096);
  float*  Lws = (float*)(ws + 22118400);
  float*  vec = (float*)(ws + 22216704);

  float* out  = (float*)d_out;
  float* prob = out + (size_t)NB_ * QL_ * EMB_;

  dim3 blk(256);
  gemm_rowmajor<0><<<dim3(16, 6), blk, 0, stream>>>(Q,   Wq, bq, qf, nullptr);
  gemm_rowmajor<1><<<dim3(32, 6), blk, 0, stream>>>(K,   Wk, bk, nullptr, kbf);
  gemm_rowmajor<2><<<dim3(32, 6), blk, 0, stream>>>(V,   Wv, bv, nullptr, vtb);
  gemm_rowmajor<1><<<dim3(16, 6), blk, 0, stream>>>(pos, Wr, nullptr, nullptr, rbf);

  attn_stats<<<dim3(32, NH_, NB_), blk, 0, stream>>>(qf, kbf, rbf, rwb, rrb,
                                                     amask, Mws, Lws);
  attn_pv<<<dim3(32, NH_, NB_), blk, 0, stream>>>(qf, kbf, rbf, vtb, rwb, rrb,
                                                  amask, Mws, Lws, prob, vec);

  gemm_rowmajor<0><<<dim3(16, 6), blk, 0, stream>>>(vec, Wo, bo, out, nullptr);
}
